// XPUInt4Linear_77292231459026
// MI455X (gfx1250) — compile-verified, matrix operand order checked by reading, not run on hardware
//
#include <hip/hip_runtime.h>

typedef __attribute__((ext_vector_type(16))) _Float16 v16h;
typedef __attribute__((ext_vector_type(8)))  _Float16 v8h;
typedef __attribute__((ext_vector_type(4)))  _Float16 v4h;
typedef __attribute__((ext_vector_type(2)))  _Float16 v2h;
typedef __attribute__((ext_vector_type(8)))  float    v8f;
typedef __attribute__((ext_vector_type(4)))  int      v4i;

typedef __attribute__((address_space(1))) v4i gv4i;   // global
typedef __attribute__((address_space(3))) v4i lv4i;   // LDS

#define K_FEAT   4096
#define N_FEAT   4096
#define QW_COLS  (K_FEAT / 8)     // 512 packed i32 per output row
#define GROUPS   (K_FEAT / 128)   // 32 groups
#define QZ_COLS  (GROUPS / 8)     // 4 packed i32 of zeros per row

#define BM 128
#define BN 256
#define BK 128
#define LDH 136                   // padded LDS stride in halves
#define NCHUNK (K_FEAT / BK)      // 32
#define STAGE  ((BM + BN) * LDH)  // halves per pipeline stage (2 stages ~ 204 KB LDS)

#define HAS_ASYNC_LDS __has_builtin(__builtin_amdgcn_global_load_async_to_lds_b128)

// Copy 16 bytes global -> LDS. Async (ASYNCcnt-tracked) on toolchains that expose
// the gfx1250 builtin; plain load+ds_store otherwise.
__device__ __forceinline__ void copy_b128_to_lds(const _Float16* g, _Float16* l) {
#if HAS_ASYNC_LDS
    __builtin_amdgcn_global_load_async_to_lds_b128((gv4i*)g, (lv4i*)l, 0, 0);
#else
    *(v8h*)l = *(const v8h*)g;
#endif
}

__device__ __forceinline__ void wait_async_lds() {
#if __has_builtin(__builtin_amdgcn_s_wait_asynccnt)
    __builtin_amdgcn_s_wait_asynccnt(0);
#elif HAS_ASYNC_LDS
    asm volatile("s_wait_asynccnt 0" ::: "memory");
#endif
}

// Prepass: x f32 -> f16, with (j, j+4) interleave inside every aligned group of 8
// K-values. This matches the natural nibble-pair order of the int4 dequant, so the
// GEMM needs no reorder shuffles (K-permutation applied to both A and B is a no-op
// for the WMMA dot product).
__global__ __launch_bounds__(256)
void convert_x_f16(const float* __restrict__ x, _Float16* __restrict__ xh)
{
    const size_t i = (size_t)blockIdx.x * 256 + threadIdx.x;   // one group of 8
    const float4 f0 = *(const float4*)&x[i * 8];
    const float4 f1 = *(const float4*)&x[i * 8 + 4];
    v8h h;
    h[0] = (_Float16)f0.x; h[1] = (_Float16)f1.x;
    h[2] = (_Float16)f0.y; h[3] = (_Float16)f1.y;
    h[4] = (_Float16)f0.z; h[5] = (_Float16)f1.z;
    h[6] = (_Float16)f0.w; h[7] = (_Float16)f1.w;
    *(v8h*)&xh[i * 8] = h;
}

template <bool USE_XH>
__global__ __launch_bounds__(256)
void int4_gemm_wmma_f16(const float*    __restrict__ x,
                        const _Float16* __restrict__ xh,
                        const int*      __restrict__ qweight,
                        const int*      __restrict__ qzeros,
                        const float*    __restrict__ scales,
                        const float*    __restrict__ bias,
                        float*          __restrict__ out)
{
    extern __shared__ _Float16 smem[];   // 2 x (As[BM][LDH] + Bs[BN][LDH])

    const int tid      = threadIdx.x;
    const int lane     = tid & 31;
    const int wave     = tid >> 5;       // 0..7
    const int wm       = wave & 1;       // M half of 2x4 wave grid
    const int wn       = wave >> 1;      // N quarter (64 cols each)
    const int laneM    = lane & 15;
    const int laneHalf = lane >> 4;

    const int m0 = blockIdx.y * BM;
    const int n0 = blockIdx.x * BN;

    // weight dequant: 1 thread per output row, 16 packed i32 (128 K) each
    const int gn = n0 + tid;

    v8f acc[4][4];
    #pragma unroll
    for (int i = 0; i < 4; ++i)
        #pragma unroll
        for (int j = 0; j < 4; ++j)
            #pragma unroll
            for (int r = 0; r < 8; ++r)
                acc[i][j][r] = 0.0f;

    int4  qs[4];
    float zf, sf;

    // ---- A tile -> LDS (async on the fast path; overlapped with WMMA) ----
    auto copy_a = [&](int kb, int buf) {
        const int k0 = kb * BK;
        _Float16* As = smem + buf * STAGE;
        #pragma unroll
        for (int i = 0; i < 8; ++i) {
            const int pidx = tid + i * 256;        // 2048 groups of 8 halves
            const int row  = pidx >> 4;            // 16 groups per row
            const int p    = pidx & 15;
            _Float16* l = &As[row * LDH + p * 8];
            if (USE_XH) {
                copy_b128_to_lds(&xh[(size_t)(m0 + row) * K_FEAT + k0 + p * 8], l);
            } else {
                const float4 f0 = *(const float4*)&x[(size_t)(m0 + row) * K_FEAT + k0 + p * 8];
                const float4 f1 = *(const float4*)&x[(size_t)(m0 + row) * K_FEAT + k0 + p * 8 + 4];
                v8h h;
                h[0] = (_Float16)f0.x; h[1] = (_Float16)f1.x;
                h[2] = (_Float16)f0.y; h[3] = (_Float16)f1.y;
                h[4] = (_Float16)f0.z; h[5] = (_Float16)f1.z;
                h[6] = (_Float16)f0.w; h[7] = (_Float16)f1.w;
                *(v8h*)l = h;
            }
        }
    };

    auto load_q = [&](int kb) {
        const int g = kb;                          // group == K-chunk
        const unsigned zq =
            ((const unsigned*)qzeros)[(size_t)gn * QZ_COLS + (g >> 3)];
        zf = (float)((zq >> ((g & 7) * 4)) & 15u);
        sf = scales[(size_t)gn * GROUPS + g];
        const int* qp = &qweight[(size_t)gn * QW_COLS + (kb * BK >> 3)];
        #pragma unroll
        for (int i = 0; i < 4; ++i)
            qs[i] = *(const int4*)(qp + i * 4);
    };

    // ---- dequant 16 i32 -> 128 f16 weights via exact packed-f16 math ----
    auto store_b = [&](int buf) {
        _Float16* Bs = smem + buf * STAGE + BM * LDH;
        const _Float16 sh = (_Float16)sf;
        const _Float16 ch = (_Float16)(-(1024.0f + zf));   // integer <= 1039: exact in f16
        const v2h s2 = {sh, sh};
        const v2h c2 = {ch, ch};
        #pragma unroll
        for (int i = 0; i < 4; ++i) {
            const int qv[4] = {qs[i].x, qs[i].y, qs[i].z, qs[i].w};
            #pragma unroll
            for (int j = 0; j < 4; ++j) {
                const unsigned q = (unsigned)qv[j];
                // (q>>4t) & 0x000F000F | 0x64006400  ==  f16 pair (1024+n_t, 1024+n_{t+4})
                const unsigned u0 = ( q        & 0x000F000Fu) | 0x64006400u;
                const unsigned u1 = ((q >> 4)  & 0x000F000Fu) | 0x64006400u;
                const unsigned u2 = ((q >> 8)  & 0x000F000Fu) | 0x64006400u;
                const unsigned u3 = ((q >> 12) & 0x000F000Fu) | 0x64006400u;
                const v2h w0 = (__builtin_bit_cast(v2h, u0) + c2) * s2;  // v_pk_add/mul_f16
                const v2h w1 = (__builtin_bit_cast(v2h, u1) + c2) * s2;
                const v2h w2 = (__builtin_bit_cast(v2h, u2) + c2) * s2;
                const v2h w3 = (__builtin_bit_cast(v2h, u3) + c2) * s2;
                const v4h lo = __builtin_shufflevector(w0, w1, 0, 1, 2, 3);
                const v4h hi = __builtin_shufflevector(w2, w3, 0, 1, 2, 3);
                const v8h h  = __builtin_shufflevector(lo, hi, 0, 1, 2, 3, 4, 5, 6, 7);
                *(v8h*)&Bs[tid * LDH + (i * 4 + j) * 8] = h;   // ds_store_b128
            }
        }
    };

    auto compute = [&](int buf) {
        const _Float16* As = smem + buf * STAGE;
        const _Float16* Bs = As + BM * LDH;
        #pragma unroll
        for (int ks = 0; ks < 4; ++ks) {
            const int kk = ks * 32 + (laneHalf ? 8 : 0);
            v16h a[4], b[4];
            #pragma unroll
            for (int i = 0; i < 4; ++i) {
                const _Float16* p = &As[(wm * 64 + i * 16 + laneM) * LDH + kk];
                const v8h lo = *(const v8h*)p;
                const v8h hi = *(const v8h*)(p + 16);
                a[i] = __builtin_shufflevector(lo, hi, 0,1,2,3,4,5,6,7,8,9,10,11,12,13,14,15);
            }
            #pragma unroll
            for (int j = 0; j < 4; ++j) {
                const _Float16* p = &Bs[(wn * 64 + j * 16 + laneM) * LDH + kk];
                const v8h lo = *(const v8h*)p;
                const v8h hi = *(const v8h*)(p + 16);
                b[j] = __builtin_shufflevector(lo, hi, 0,1,2,3,4,5,6,7,8,9,10,11,12,13,14,15);
            }
            #pragma unroll
            for (int i = 0; i < 4; ++i)
                #pragma unroll
                for (int j = 0; j < 4; ++j)
                    acc[i][j] = __builtin_amdgcn_wmma_f32_16x16x32_f16(
                        false, a[i], false, b[j], (short)0, acc[i][j], false, false);
        }
    };

    // ---- pipelined main loop ----
    load_q(0);
    copy_a(0, 0);
    store_b(0);
    wait_async_lds();
    __syncthreads();

    for (int kb = 0; kb < NCHUNK; ++kb) {
        const int cur = kb & 1;
        if (kb + 1 < NCHUNK) {
            load_q(kb + 1);              // global loads, consumed after compute
            copy_a(kb + 1, cur ^ 1);     // async A fill overlaps the WMMA block
        }
        compute(cur);
        if (kb + 1 < NCHUNK)
            store_b(cur ^ 1);            // dequant + ds_store into the other buffer
        wait_async_lds();
        __syncthreads();
    }

    // ---- epilogue: bias add + f32 store ----
    #pragma unroll
    for (int j = 0; j < 4; ++j) {
        const int n    = n0 + wn * 64 + j * 16 + laneM;
        const float bv = bias[n];
        #pragma unroll
        for (int i = 0; i < 4; ++i) {
            const int mbase = m0 + wm * 64 + i * 16 + laneHalf * 8;
            #pragma unroll
            for (int r = 0; r < 8; ++r)
                out[(size_t)(mbase + r) * N_FEAT + n] = acc[i][j][r] + bv;
        }
    }
}

extern "C" void kernel_launch(void* const* d_in, const int* in_sizes, int n_in,
                              void* d_out, int out_size, void* d_ws, size_t ws_size,
                              hipStream_t stream) {
    const float* x       = (const float*)d_in[0];
    const int*   qweight = (const int*)d_in[1];
    const int*   qzeros  = (const int*)d_in[2];
    const float* scales  = (const float*)d_in[3];
    const float* bias    = (const float*)d_in[4];
    float*       out     = (float*)d_out;

    const int    M         = in_sizes[0] / K_FEAT;                  // 4096 tokens
    const size_t xh_bytes  = (size_t)M * K_FEAT * sizeof(_Float16); // 32 MB
    dim3 grid(N_FEAT / BN, M / BM);                                 // 16 x 32
    const size_t lds_bytes = (size_t)2 * STAGE * sizeof(_Float16);  // ~204 KB (320 KB WGP)

    if (ws_size >= xh_bytes) {
        _Float16* xh = (_Float16*)d_ws;
        const size_t groups8 = (size_t)M * K_FEAT / 8;
        convert_x_f16<<<dim3(groups8 / 256), 256, 0, stream>>>(x, xh);
        int4_gemm_wmma_f16<true><<<grid, 256, lds_bytes, stream>>>(
            x, xh, qweight, qzeros, scales, bias, out);
    } else {
        int4_gemm_wmma_f16<false><<<grid, 256, lds_bytes, stream>>>(
            x, nullptr, qweight, qzeros, scales, bias, out);
    }
}
